// MultilayerLSTM_2843268350650
// MI455X (gfx1250) — matrix operand order repaired and checked by reading, not compiled, not run on hardware
//
#include <hip/hip_runtime.h>

// ---------------------------------------------------------------------------
// 2-layer LSTM scan for MI455X (gfx1250, wave32, WMMA bf16 16x16x32).
//
// Strategy (latency-bound problem: 1024 sequential steps):
//  * 32 persistent workgroups, each owns 16 batch rows; h/c live on-chip.
//  * Both layers fused per timestep (out0 never hits global memory).
//  * W_hh0/W_hh1 pre-swizzled to bf16 WMMA B-fragments, resident in LDS
//    (256 KB of the 320 KB WGP LDS). W_ih1 fragments streamed from L2.
//  * Wave w owns n-tiles {w, w+8, w+16, w+24} => its 4 accumulators are
//    gates i,f,g,o for hidden slice [16w,16w+16) at identical lane/vgpr
//    positions -> elementwise LSTM update entirely in registers.
// ---------------------------------------------------------------------------

#define N_FEAT 6
#define N_HID  128
#define SEQ    1024

typedef __attribute__((ext_vector_type(16))) __bf16        v16bf;
typedef __attribute__((ext_vector_type(8)))  float         v8f;
typedef __attribute__((ext_vector_type(4)))  unsigned int  v4u;

union FragU { v16bf v; v4u q[2]; };

static __device__ __forceinline__ unsigned short f2bf(float x) {
  unsigned int u = __float_as_uint(x);
  u += 0x7FFFu + ((u >> 16) & 1u);          // round-to-nearest-even
  return (unsigned short)(u >> 16);
}
static __device__ __forceinline__ float bf2f(unsigned short b) {
  return __uint_as_float(((unsigned int)b) << 16);
}
static __device__ __forceinline__ float fast_sigmoid(float x) {
  float e = __builtin_amdgcn_exp2f(-1.44269504f * x);   // v_exp_f32
  return __builtin_amdgcn_rcpf(1.0f + e);               // v_rcp_f32
}
static __device__ __forceinline__ float fast_tanh(float x) {
  float e = __builtin_amdgcn_exp2f(2.88539008f * x);    // 2^(2x*log2e) = e^(2x)
  return 1.0f - 2.0f * __builtin_amdgcn_rcpf(e + 1.0f);
}

// B-fragment load from chunked layout: [frag][chunk(2)][lane(32)][8 halves].
// Per-lane: two contiguous 16B reads (2-phase bank access in LDS).
static __device__ __forceinline__ v16bf load_frag(const unsigned short* p, int f, int lane) {
  FragU u;
  u.q[0] = *(const v4u*)(p + ((f * 2 + 0) * 32 + lane) * 8);
  u.q[1] = *(const v4u*)(p + ((f * 2 + 1) * 32 + lane) * 8);
  return u.v;
}

// A-fragment (16-bit 16x32): lane L holds row m=L&15;
// 16 contiguous elements at K = kt*32 + 16*half (matches B fragments).
static __device__ __forceinline__ v16bf load_A(const unsigned short* row0, int strideH,
                                               int kt, int lane) {
  int m = lane & 15, half = lane >> 4;
  const unsigned short* r = row0 + m * strideH + kt * 32 + half * 16;
  FragU u;
  u.q[0] = *(const v4u*)(r);
  u.q[1] = *(const v4u*)(r + 8);
  return u.v;
}

#define SH 136                 // h row stride in halves (128 + 16B pad -> 2-phase LDS)
#define SX 40                  // x row stride in halves (32 + pad)
#define SMEM_BYTES ((131072 * 2) + (16 * SH * 2) * 2 + (16 * SX * 2))   // 272128

// ---------------------------------------------------------------------------
// Prep: swizzle fp32 weight [4H, K] into bf16 WMMA B-fragments.
// Fragment f = nt*KT + kt; lane = column n within n-tile; per-lane halves are
// K-contiguous per half-wave (k = kt*32 + 16*half + e; same order as load_A).
// ---------------------------------------------------------------------------
__global__ void prep_frags(const float* __restrict__ W, int Kact, int ldW, int KT,
                           unsigned short* __restrict__ dst) {
  int idx = blockIdx.x * blockDim.x + threadIdx.x;
  int total = 32 * KT * 32 * 16;
  if (idx >= total) return;
  int e    = idx & 15;
  int lane = (idx >> 4) & 31;
  int f    = idx >> 9;
  int kt   = f % KT;
  int nt   = f / KT;
  int n    = nt * 16 + (lane & 15);
  int half = lane >> 4;
  int k    = kt * 32 + half * 16 + e;
  float v  = (k < Kact) ? W[(size_t)n * ldW + k] : 0.0f;
  int chunk = e >> 3, within = e & 7;
  dst[((size_t)(f * 2 + chunk) * 32 + lane) * 8 + within] = f2bf(v);
}

__global__ void prep_bias(const float* __restrict__ bi0, const float* __restrict__ bh0,
                          const float* __restrict__ bi1, const float* __restrict__ bh1,
                          float* __restrict__ d0, float* __restrict__ d1) {
  int i = blockIdx.x * blockDim.x + threadIdx.x;
  if (i < 512) d0[i] = bi0[i] + bh0[i];
  else if (i < 1024) { int k = i - 512; d1[k] = bi1[k] + bh1[k]; }
}

// ---------------------------------------------------------------------------
// Persistent fused 2-layer LSTM scan + final linear.
// ---------------------------------------------------------------------------
__global__ __launch_bounds__(256) void lstm2_scan(
    const float* __restrict__ X,
    const unsigned short* __restrict__ wsWhhBoth,   // whh0 frags | whh1 frags
    const unsigned short* __restrict__ wsWih1,      // streamed from L2 each step
    const unsigned short* __restrict__ wsWx0,
    const float* __restrict__ bias0,
    const float* __restrict__ bias1,
    const float* __restrict__ Wlin,
    const float* __restrict__ blin,
    float* __restrict__ out) {
  extern __shared__ unsigned short smem[];
  unsigned short* whh0 = smem;                       // 65536 halves (128 KB)
  unsigned short* whh1 = smem + 65536;               // 65536 halves
  unsigned short* h0   = smem + 131072;              // 16*SH
  unsigned short* h1   = h0 + 16 * SH;
  unsigned short* xls  = h1 + 16 * SH;               // 16*SX

  const int tid   = threadIdx.x;
  const int lane  = tid & 31;
  const int w     = tid >> 5;          // wave 0..7 -> hidden slice [16w,16w+16)
  const int nl    = lane & 15;
  const int half  = lane >> 4;
  const int j     = 16 * w + nl;       // hidden index owned by this lane
  const int bbase = blockIdx.x * 16;   // batch rows [bbase, bbase+16)

  // One-time LDS fill: 256 KB of recurrent-weight fragments.
  {
    const v4u* src = (const v4u*)wsWhhBoth;
    v4u* dst = (v4u*)smem;
    for (int i = tid; i < (131072 * 2) / 16; i += 256) dst[i] = src[i];
  }
  for (int i = tid; i < 16 * SH * 2 + 16 * SX; i += 256) h0[i] = 0;  // h0|h1|x zero
  __syncthreads();

  // Hoisted constants: W_ih0 fragments + fused biases live in registers.
  v16bf bx0[4];
  float b0r[4], b1r[4];
#pragma unroll
  for (int q = 0; q < 4; ++q) {
    bx0[q] = load_frag(wsWx0, w + 8 * q, lane);
    b0r[q] = bias0[q * N_HID + j];
    b1r[q] = bias1[q * N_HID + j];
  }
  float c0r[8], c1r[8];
#pragma unroll
  for (int v = 0; v < 8; ++v) { c0r[v] = 0.0f; c1r[v] = 0.0f; }

  const int sm = tid / 6, sd = tid - sm * 6;
  const bool stager = tid < 96;
  const float* Xs = X + (size_t)(bbase + sm) * SEQ * N_FEAT + sd;

  for (int t = 0; t < SEQ; ++t) {
    if (stager) {
      xls[sm * SX + sd] = f2bf(Xs[(size_t)t * N_FEAT]);
      if (t + 1 < SEQ) __builtin_prefetch(&Xs[(size_t)(t + 1) * N_FEAT], 0, 0);
    }
    __syncthreads();

    // ================= layer 0: gates = b0 + x W_ih0^T + h0 W_hh0^T ========
    v8f acc[4];
#pragma unroll
    for (int q = 0; q < 4; ++q) {
#pragma unroll
      for (int i = 0; i < 8; ++i) acc[q][i] = b0r[q];
    }
    {
      v16bf ax = load_A(xls, SX, 0, lane);
#pragma unroll
      for (int q = 0; q < 4; ++q)
        acc[q] = __builtin_amdgcn_wmma_f32_16x16x32_bf16(false, ax, false, bx0[q],
                                                         (short)0, acc[q], false, false);
    }
#pragma unroll
    for (int kt = 0; kt < 4; ++kt) {
      v16bf a = load_A(h0, SH, kt, lane);
#pragma unroll
      for (int q = 0; q < 4; ++q) {
        v16bf b = load_frag(whh0, (w + 8 * q) * 4 + kt, lane);
        acc[q] = __builtin_amdgcn_wmma_f32_16x16x32_bf16(false, a, false, b,
                                                         (short)0, acc[q], false, false);
      }
    }
    // acc[0..3] = i,f,g,o for (m = vgpr+8*half, hidden j) -- all local.
    unsigned short hnew0[8];
#pragma unroll
    for (int v = 0; v < 8; ++v) {
      float ig = fast_sigmoid(acc[0][v]);
      float fg = fast_sigmoid(acc[1][v]);
      float gg = fast_tanh(acc[2][v]);
      float og = fast_sigmoid(acc[3][v]);
      float c  = fg * c0r[v] + ig * gg;
      c0r[v]   = c;
      hnew0[v] = f2bf(og * fast_tanh(c));
    }
    __syncthreads();                       // all reads of old h0 complete
#pragma unroll
    for (int v = 0; v < 8; ++v) h0[(v + 8 * half) * SH + j] = hnew0[v];
    __syncthreads();                       // h0 now holds out0_t

    // ================= layer 1: gates = b1 + h0 W_ih1^T + h1 W_hh1^T =======
#pragma unroll
    for (int q = 0; q < 4; ++q) {
#pragma unroll
      for (int i = 0; i < 8; ++i) acc[q][i] = b1r[q];
    }
#pragma unroll
    for (int kt = 0; kt < 4; ++kt) {
      v16bf a0 = load_A(h0, SH, kt, lane);
#pragma unroll
      for (int q = 0; q < 4; ++q) {
        v16bf b = load_frag(wsWih1, (w + 8 * q) * 4 + kt, lane);   // L2-resident
        acc[q] = __builtin_amdgcn_wmma_f32_16x16x32_bf16(false, a0, false, b,
                                                         (short)0, acc[q], false, false);
      }
      v16bf a1 = load_A(h1, SH, kt, lane);
#pragma unroll
      for (int q = 0; q < 4; ++q) {
        v16bf b = load_frag(whh1, (w + 8 * q) * 4 + kt, lane);
        acc[q] = __builtin_amdgcn_wmma_f32_16x16x32_bf16(false, a1, false, b,
                                                         (short)0, acc[q], false, false);
      }
    }
    unsigned short hnew1[8];
#pragma unroll
    for (int v = 0; v < 8; ++v) {
      float ig = fast_sigmoid(acc[0][v]);
      float fg = fast_sigmoid(acc[1][v]);
      float gg = fast_tanh(acc[2][v]);
      float og = fast_sigmoid(acc[3][v]);
      float c  = fg * c1r[v] + ig * gg;
      c1r[v]   = c;
      hnew1[v] = f2bf(og * fast_tanh(c));
    }
    __syncthreads();                       // all reads of old h1 complete
#pragma unroll
    for (int v = 0; v < 8; ++v) h1[(v + 8 * half) * SH + j] = hnew1[v];
    // next iteration's post-stage barrier separates these writes from reads
  }
  __syncthreads();

  // Final linear: out[m,f] = h1[m,:] . W_lin[f,:] + b_lin[f]  (16x6 per WG)
  if (tid < 96) {
    int m = tid / 6, ff = tid - m * 6;
    float s = blin[ff];
    for (int k = 0; k < N_HID; ++k) s += bf2f(h1[m * SH + k]) * Wlin[ff * N_HID + k];
    out[(size_t)(bbase + m) * N_FEAT + ff] = s;
  }
}

// ---------------------------------------------------------------------------
extern "C" void kernel_launch(void* const* d_in, const int* in_sizes, int n_in,
                              void* d_out, int out_size, void* d_ws, size_t ws_size,
                              hipStream_t stream) {
  (void)in_sizes; (void)n_in; (void)out_size; (void)ws_size;

  const float* X     = (const float*)d_in[0];
  const float* W_ih0 = (const float*)d_in[1];
  const float* W_hh0 = (const float*)d_in[2];
  const float* b_ih0 = (const float*)d_in[3];
  const float* b_hh0 = (const float*)d_in[4];
  const float* W_ih1 = (const float*)d_in[5];
  const float* W_hh1 = (const float*)d_in[6];
  const float* b_ih1 = (const float*)d_in[7];
  const float* b_hh1 = (const float*)d_in[8];
  const float* W_lin = (const float*)d_in[9];
  const float* b_lin = (const float*)d_in[10];

  // Workspace layout (halves): whh0 | whh1 | wih1 | wx0 | bias0/1 (f32).
  unsigned short* ws    = (unsigned short*)d_ws;
  unsigned short* whh0F = ws;                    // 65536 halves
  unsigned short* whh1F = ws + 65536;            // 65536
  unsigned short* wih1F = ws + 131072;           // 65536
  unsigned short* wx0F  = ws + 196608;           // 16384
  float* bias0 = (float*)(ws + 212992);          // 512 f32, 16B aligned
  float* bias1 = bias0 + 512;                    // 512 f32  (~430 KB total)

  prep_frags<<<256, 256, 0, stream>>>(W_hh0, 128, 128, 4, whh0F);
  prep_frags<<<256, 256, 0, stream>>>(W_hh1, 128, 128, 4, whh1F);
  prep_frags<<<256, 256, 0, stream>>>(W_ih1, 128, 128, 4, wih1F);
  prep_frags<<<64,  256, 0, stream>>>(W_ih0, 6, 6, 1, wx0F);
  prep_bias <<<4,   256, 0, stream>>>(b_ih0, b_hh0, b_ih1, b_hh1, bias0, bias1);

  hipFuncSetAttribute(reinterpret_cast<const void*>(lstm2_scan),
                      hipFuncAttributeMaxDynamicSharedMemorySize, SMEM_BYTES);
  lstm2_scan<<<32, 256, SMEM_BYTES, stream>>>(X, whh0F, wih1F, wx0F,
                                              bias0, bias1, W_lin, b_lin,
                                              (float*)d_out);
}
